// GenericMoeLayer_20358144983695
// MI455X (gfx1250) — matrix operand order repaired and checked by reading, not compiled, not run on hardware
//
#include <hip/hip_runtime.h>
#include <hip/hip_bf16.h>

typedef __attribute__((ext_vector_type(16))) __bf16 v16bf;
typedef __attribute__((ext_vector_type(8)))  __bf16 v8bf;
typedef __attribute__((ext_vector_type(4)))  __bf16 v4bf;
typedef __attribute__((ext_vector_type(8)))  float  v8f;

#define T_TOK 1024
#define H_DIM 1024
#define F_DIM 768
#define E_NUM 32
#define K_TOP 4
#define MT    8                                // M-tiles (of 16 rows) per wave
#define PADCAP (T_TOK * K_TOP + E_NUM * 16)    // 4608 padded slots max

// ---------------- ws layout (bytes) ----------------
#define WS_COUNTS   0                              // E ints
#define WS_CURSOR   256                            // E ints
#define WS_OFFS     512                            // E ints (padded exclusive scan)
#define WS_TOKE     1024                           // T*K ints   (16384 B)
#define WS_TOKW     (WS_TOKE + 16384)              // T*K floats
#define WS_TOKSLOT  (WS_TOKW + 16384)              // T*K ints
#define WS_SLOTTOK  (WS_TOKSLOT + 16384)           // PADCAP ints   (18432 B)
#define WS_SLOTW    (WS_SLOTTOK + 18432)           // PADCAP floats
#define WS_ACT      (WS_SLOTW + 18432)             // PADCAP*F bf16 (7077888 B)
#define WS_Y        (WS_ACT + PADCAP * F_DIM * 2)  // PADCAP*H floats (18874368 B)
#define WS_HSB      (WS_Y + PADCAP * H_DIM * 4)    // T*H bf16 (2 MB)

// ---------------- kernel 0: zero per-call atomics ----------------
__global__ void moe_zero(int* __restrict__ counts, int* __restrict__ cursor) {
    int i = threadIdx.x;
    if (i < E_NUM) { counts[i] = 0; cursor[i] = 0; }
}

// ---------------- kernel 0b: hs fp32 -> bf16 (one-time, 6 MB traffic) --------
__global__ void moe_cvt_hs(const float* __restrict__ hs, __bf16* __restrict__ hsb) {
    int i = blockIdx.x * blockDim.x + threadIdx.x;      // one float4 per thread
    if (i >= T_TOK * H_DIM / 4) return;
    float4 v = *(const float4*)(hs + 4 * (size_t)i);
    v4bf o;
    o[0] = (__bf16)v.x; o[1] = (__bf16)v.y; o[2] = (__bf16)v.z; o[3] = (__bf16)v.w;
    *(v4bf*)(hsb + 4 * (size_t)i) = o;
}

// ---------------- kernel 1: router (1 wave per token, lane = expert) ----------
__global__ __launch_bounds__(32) void moe_router(const float* __restrict__ hs,
                                                 const float* __restrict__ gw,
                                                 int*   __restrict__ counts,
                                                 int*   __restrict__ tok_e,
                                                 float* __restrict__ tok_w) {
    const int t    = blockIdx.x;
    const int lane = threadIdx.x;          // expert id (E == wave32)
    const float4* hrow = (const float4*)(hs + (size_t)t * H_DIM);
    const float4* grow = (const float4*)(gw + (size_t)lane * H_DIM);
    float acc = 0.f;
    #pragma unroll 4
    for (int i = 0; i < H_DIM / 4; ++i) {
        float4 a = hrow[i], b = grow[i];
        acc += a.x * b.x + a.y * b.y + a.z * b.z + a.w * b.w;
    }
    // softmax across the 32 lanes
    float m = acc;
    for (int off = 16; off; off >>= 1) m = fmaxf(m, __shfl_xor(m, off, 32));
    float p = __expf(acc - m);
    float s = p;
    for (int off = 16; off; off >>= 1) s += __shfl_xor(s, off, 32);
    p /= s;
    // top-4 via repeated wave argmax (all lanes converge to same winner)
    float myp = p, wsum = 0.f;
    int   my_e = 0; float my_p = 0.f;
    for (int k = 0; k < K_TOP; ++k) {
        float bv = myp; int bl = lane;
        for (int off = 16; off; off >>= 1) {
            float ov = __shfl_xor(bv, off, 32);
            int   ol = __shfl_xor(bl, off, 32);
            if (ov > bv || (ov == bv && ol < bl)) { bv = ov; bl = ol; }
        }
        wsum += bv;
        if (lane == k)  { my_e = bl; my_p = bv; }
        if (lane == bl) { myp = -1.f; }
    }
    if (lane < K_TOP) {
        tok_e[t * K_TOP + lane] = my_e;
        tok_w[t * K_TOP + lane] = my_p / wsum;
        atomicAdd(&counts[my_e], 1);
    }
}

// ---------------- kernel 2: padded exclusive scan of counts (E=32, trivial) --
__global__ void moe_offsets(const int* __restrict__ counts, int* __restrict__ offs) {
    if (threadIdx.x == 0) {
        int acc = 0;
        for (int e = 0; e < E_NUM; ++e) {
            offs[e] = acc;
            acc += (counts[e] + 15) & ~15;
        }
    }
}

// ---------------- kernel 3: token -> slot assignment -------------------------
__global__ void moe_assign(const int* __restrict__ tok_e, const float* __restrict__ tok_w,
                           const int* __restrict__ offs,  int* __restrict__ cursor,
                           int* __restrict__ tok_slot, int* __restrict__ slot_tok,
                           float* __restrict__ slot_w) {
    int t = blockIdx.x * blockDim.x + threadIdx.x;
    if (t >= T_TOK) return;
    for (int k = 0; k < K_TOP; ++k) {
        int e    = tok_e[t * K_TOP + k];
        int pos  = atomicAdd(&cursor[e], 1);
        int slot = offs[e] + pos;
        tok_slot[t * K_TOP + k] = slot;
        slot_tok[slot] = t;
        slot_w[slot]   = tok_w[t * K_TOP + k];
    }
}

// A fragment from a bf16 row: element s <-> k = kb + hi*16 + s
__device__ __forceinline__ v16bf load_a_frag(const __bf16* row, int kb, int hi) {
    v8bf lo = *(const v8bf*)(row + kb + hi * 16);
    v8bf hh = *(const v8bf*)(row + kb + hi * 16 + 8);
    return __builtin_shufflevector(lo, hh, 0, 1, 2, 3, 4, 5, 6, 7,
                                           8, 9, 10, 11, 12, 13, 14, 15);
}

// ---------------- kernel 4: grouped GEMM1 + SiLU-gate (bf16 WMMA) ------------
// one wave computes MT 16x16 token tiles against one 16-col slice of gate AND up;
// B fragments are loaded once per K-step and reused across the MT tiles.
__global__ __launch_bounds__(32) void moe_gemm1(
    const __bf16* __restrict__ hsb, const float* __restrict__ w1,
    const int* __restrict__ counts, const int* __restrict__ offs,
    const int* __restrict__ slot_tok, const float* __restrict__ slot_w,
    __bf16* __restrict__ act) {
    const int e   = blockIdx.z;
    const int mg  = blockIdx.y;               // group of MT m-tiles
    const int ft  = blockIdx.x;
    const int cnt = counts[e];
    if (mg * (MT * 16) >= cnt) return;        // uniform per-wave exit
    const int base_slot = offs[e] + mg * (MT * 16);
    const int lane = threadIdx.x;
    const int r = lane & 15, hi = lane >> 4;

    // per-lane bf16 token rows for each of the MT tiles (pad rows -> token 0)
    const __bf16* arows[MT];
    #pragma unroll
    for (int mi = 0; mi < MT; ++mi) {
        int row_in = mg * (MT * 16) + mi * 16 + r;
        int tok = (row_in < cnt) ? slot_tok[base_slot + mi * 16 + r] : 0;
        arows[mi] = hsb + (size_t)tok * H_DIM;
    }

    const int fb = ft * 16;
    const float* bg = w1 + (size_t)e * 2 * F_DIM * H_DIM + (size_t)(fb + r) * H_DIM;
    const float* bu = bg + (size_t)F_DIM * H_DIM;

    v8f accg[MT] = {}, accu[MT] = {};
    for (int kb = 0; kb < H_DIM; kb += 32) {
        const int kw = kb + hi * 16;          // B: lanes 0-15 K 0-15, 16-31 K 16-31
        __builtin_prefetch(bg + kw + 128, 0, 0);
        __builtin_prefetch(bu + kw + 128, 0, 0);
        v16bf b0, b1;
        #pragma unroll
        for (int j = 0; j < 4; ++j) {
            float4 g4 = *(const float4*)(bg + kw + 4 * j);
            float4 u4 = *(const float4*)(bu + kw + 4 * j);
            b0[4 * j] = (__bf16)g4.x; b0[4 * j + 1] = (__bf16)g4.y;
            b0[4 * j + 2] = (__bf16)g4.z; b0[4 * j + 3] = (__bf16)g4.w;
            b1[4 * j] = (__bf16)u4.x; b1[4 * j + 1] = (__bf16)u4.y;
            b1[4 * j + 2] = (__bf16)u4.z; b1[4 * j + 3] = (__bf16)u4.w;
        }
        #pragma unroll
        for (int mi = 0; mi < MT; ++mi) {
            v16bf a = load_a_frag(arows[mi], kb, hi);
            accg[mi] = __builtin_amdgcn_wmma_f32_16x16x32_bf16(
                false, a, false, b0, (short)0, accg[mi], false, false);
            accu[mi] = __builtin_amdgcn_wmma_f32_16x16x32_bf16(
                false, a, false, b1, (short)0, accu[mi], false, false);
        }
    }
    // epilogue: act = cw * silu(g) * u ; pad rows store exact 0 (cw==0)
    #pragma unroll
    for (int mi = 0; mi < MT; ++mi) {
        int tstart = mg * (MT * 16) + mi * 16;
        if (tstart >= cnt) break;             // tile fully outside expert region
        #pragma unroll
        for (int v = 0; v < 8; ++v) {
            int m    = v + hi * 8;
            int rowe = tstart + m;
            float cw = (rowe < cnt) ? slot_w[base_slot + mi * 16 + m] : 0.f;
            float g = accg[mi][v], u = accu[mi][v];
            float sg = g * __builtin_amdgcn_rcpf(1.f + __expf(-g));   // fast silu
            act[(size_t)(base_slot + mi * 16 + m) * F_DIM + fb + r] =
                (__bf16)(sg * u * cw);
        }
    }
}

// ---------------- kernel 5: grouped GEMM2 (bf16 WMMA, K over F) --------------
__global__ __launch_bounds__(32) void moe_gemm2(
    const float* __restrict__ w2, const __bf16* __restrict__ act,
    const int* __restrict__ counts, const int* __restrict__ offs,
    float* __restrict__ y) {
    const int e   = blockIdx.z;
    const int mg  = blockIdx.y;
    const int ht  = blockIdx.x;
    const int cnt = counts[e];
    if (mg * (MT * 16) >= cnt) return;
    const int padcnt = (cnt + 15) & ~15;
    const int base_slot = offs[e] + mg * (MT * 16);
    const int lane = threadIdx.x;
    const int r = lane & 15, hi = lane >> 4;

    // per-tile A row pointers (clamped inside this expert's padded region)
    const __bf16* arows[MT];
    #pragma unroll
    for (int mi = 0; mi < MT; ++mi) {
        int row_in = mg * (MT * 16) + mi * 16 + r;
        int slot = (row_in < padcnt) ? (base_slot + mi * 16 + r) : offs[e];
        arows[mi] = act + (size_t)slot * F_DIM;
    }

    const int hb = ht * 16;
    const float* brow = w2 + (size_t)e * H_DIM * F_DIM + (size_t)(hb + r) * F_DIM;

    v8f acc[MT] = {};
    for (int kb = 0; kb < F_DIM; kb += 32) {
        const int kw = kb + hi * 16;
        __builtin_prefetch(brow + kw + 128, 0, 0);
        v16bf b;
        #pragma unroll
        for (int j = 0; j < 4; ++j) {
            float4 w4 = *(const float4*)(brow + kw + 4 * j);
            b[4 * j] = (__bf16)w4.x; b[4 * j + 1] = (__bf16)w4.y;
            b[4 * j + 2] = (__bf16)w4.z; b[4 * j + 3] = (__bf16)w4.w;
        }
        #pragma unroll
        for (int mi = 0; mi < MT; ++mi) {
            v16bf a = load_a_frag(arows[mi], kb, hi);
            acc[mi] = __builtin_amdgcn_wmma_f32_16x16x32_bf16(
                false, a, false, b, (short)0, acc[mi], false, false);
        }
    }
    #pragma unroll
    for (int mi = 0; mi < MT; ++mi) {
        int tstart = mg * (MT * 16) + mi * 16;
        if (tstart >= cnt) break;
        #pragma unroll
        for (int v = 0; v < 8; ++v) {
            int m = v + hi * 8;
            y[(size_t)(base_slot + mi * 16 + m) * H_DIM + hb + r] = acc[mi][v];
        }
    }
}

// ---------------- kernel 6: per-token gather of K=4 expert results -----------
__global__ void moe_gather(const float* __restrict__ y, const int* __restrict__ tok_slot,
                           float* __restrict__ out) {
    int idx = blockIdx.x * blockDim.x + threadIdx.x;
    if (idx >= T_TOK * H_DIM) return;
    int t = idx >> 10;           // / H_DIM
    int h = idx & (H_DIM - 1);   // % H_DIM
    float s = 0.f;
    #pragma unroll
    for (int k = 0; k < K_TOP; ++k)
        s += y[(size_t)tok_slot[t * K_TOP + k] * H_DIM + h];
    out[idx] = s;
}

// ---------------- host launcher ----------------------------------------------
extern "C" void kernel_launch(void* const* d_in, const int* in_sizes, int n_in,
                              void* d_out, int out_size, void* d_ws, size_t ws_size,
                              hipStream_t stream) {
    const float* hs = (const float*)d_in[0];   // (T, H)
    const float* gw = (const float*)d_in[1];   // (E, H)
    const float* w1 = (const float*)d_in[2];   // (E, 2F, H)
    const float* w2 = (const float*)d_in[3];   // (E, H, F)
    float* out = (float*)d_out;                // (T, H)

    char* ws = (char*)d_ws;
    int*    counts   = (int*)   (ws + WS_COUNTS);
    int*    cursor   = (int*)   (ws + WS_CURSOR);
    int*    offs     = (int*)   (ws + WS_OFFS);
    int*    tok_e    = (int*)   (ws + WS_TOKE);
    float*  tok_w    = (float*) (ws + WS_TOKW);
    int*    tok_slot = (int*)   (ws + WS_TOKSLOT);
    int*    slot_tok = (int*)   (ws + WS_SLOTTOK);
    float*  slot_w   = (float*) (ws + WS_SLOTW);
    __bf16* act      = (__bf16*)(ws + WS_ACT);
    float*  y        = (float*) (ws + WS_Y);
    __bf16* hsb      = (__bf16*)(ws + WS_HSB);

    moe_zero<<<1, 64, 0, stream>>>(counts, cursor);
    moe_cvt_hs<<<(T_TOK * H_DIM / 4 + 255) / 256, 256, 0, stream>>>(hs, hsb);
    moe_router<<<T_TOK, 32, 0, stream>>>(hs, gw, counts, tok_e, tok_w);
    moe_offsets<<<1, 32, 0, stream>>>(counts, offs);
    moe_assign<<<(T_TOK + 255) / 256, 256, 0, stream>>>(tok_e, tok_w, offs, cursor,
                                                        tok_slot, slot_tok, slot_w);
    moe_gemm1<<<dim3(F_DIM / 16, T_TOK / 16 / MT, E_NUM), 32, 0, stream>>>(
        hsb, w1, counts, offs, slot_tok, slot_w, act);
    moe_gemm2<<<dim3(H_DIM / 16, T_TOK / 16 / MT, E_NUM), 32, 0, stream>>>(
        w2, act, counts, offs, y);
    moe_gather<<<(T_TOK * H_DIM + 255) / 256, 256, 0, stream>>>(y, tok_slot, out);
}
